// Block_84293028151812
// MI455X (gfx1250) — matrix operand order repaired and checked by reading, not compiled, not run on hardware
//
#include <hip/hip_runtime.h>
#include <hip/hip_bf16.h>

// ---------------------------------------------------------------------------
// Transformer block for MI455X (gfx1250): all GEMMs via v_wmma_f32_16x16x32_f16
// GEMM: 128x128 block tile, 8 waves (32x64 per wave), double-buffered LDS.
// ---------------------------------------------------------------------------

typedef _Float16 half_t;
typedef __attribute__((ext_vector_type(8)))  _Float16 v8h;
typedef __attribute__((ext_vector_type(16))) _Float16 v16h;
typedef __attribute__((ext_vector_type(8)))  float    v8f;

union Frag16 { v16h v; v8h h[2]; };

#define N_EMBD 1024
#define N_HEAD 16
#define HEADSZ 64
#define SEQ    2048
#define BATCH  2
#define ROWS   (BATCH * SEQ)   // 4096 tokens
#define CFF    (4 * N_EMBD)    // 4096

static __device__ __forceinline__ v8f wmma_f16(v16h a, v16h b, v8f c) {
  // (neg_a, A, neg_b, B, c_mod, C, reuse_a, reuse_b)
  return __builtin_amdgcn_wmma_f32_16x16x32_f16(false, a, false, b, (short)0, c,
                                                false, false);
}

// ---------------------------------------------------------------------------
// f32 -> f16 with transpose: W[K][N] -> WT[N][K]  (coalesced read)
// ---------------------------------------------------------------------------
__global__ __launch_bounds__(256) void cvt_transpose_kernel(
    const float* __restrict__ W, half_t* __restrict__ WT, int K, int N) {
  size_t idx = (size_t)blockIdx.x * 256 + threadIdx.x;  // over K*N, n fastest
  int n = (int)(idx % N);
  size_t k = idx / N;
  WT[(size_t)n * K + k] = (half_t)W[idx];
}

// ---------------------------------------------------------------------------
// LayerNorm (fp32 in, f16 out), one block per token row, C = 1024
// ---------------------------------------------------------------------------
__global__ __launch_bounds__(256) void ln_f16_kernel(
    const float* __restrict__ x, const float* __restrict__ gw,
    const float* __restrict__ bw, half_t* __restrict__ out) {
  __shared__ float s1[256];
  __shared__ float s2[256];
  const int row = blockIdx.x, t = threadIdx.x;
  const float* xr = x + (size_t)row * N_EMBD;
  float v[4];
  float sum = 0.f, sq = 0.f;
#pragma unroll
  for (int i = 0; i < 4; i++) {
    v[i] = xr[t + i * 256];
    sum += v[i];
    sq += v[i] * v[i];
  }
  s1[t] = sum; s2[t] = sq;
  __syncthreads();
  for (int s = 128; s > 0; s >>= 1) {
    if (t < s) { s1[t] += s1[t + s]; s2[t] += s2[t + s]; }
    __syncthreads();
  }
  const float mu   = s1[0] * (1.f / N_EMBD);
  const float var  = s2[0] * (1.f / N_EMBD) - mu * mu;
  const float rstd = rsqrtf(var + 1e-5f);
  half_t* o = out + (size_t)row * N_EMBD;
#pragma unroll
  for (int i = 0; i < 4; i++) {
    int c = t + i * 256;
    o[c] = (half_t)((v[i] - mu) * rstd * gw[c] + bw[c]);
  }
}

// ---------------------------------------------------------------------------
// V[B*T][C] (f16) -> VT[B][H][Dh][T] (f16) so P@V B-fragments are contiguous
// ---------------------------------------------------------------------------
__global__ __launch_bounds__(256) void vtranspose_kernel(
    const half_t* __restrict__ V, half_t* __restrict__ VT) {
  int idx = blockIdx.x * 256 + threadIdx.x;  // over ROWS*C = 4M
  int c  = idx & (N_EMBD - 1);
  int bt = idx >> 10;
  int t  = bt & (SEQ - 1);
  int b  = bt >> 11;
  int h  = c >> 6;
  int d  = c & 63;
  VT[((((size_t)(b * N_HEAD + h)) * HEADSZ + d) << 11) + t] = V[idx];
}

// ---------------------------------------------------------------------------
// Tiled WMMA GEMM:  C[M][N] = A[M][K](f16) * BT[N][K]^T (f16)  (+bias,+relu,+resid)
// Block = 256 threads (8 waves). Block tile 128x128; wave tile 32x64
// (waves in a 4x2 grid). Double-buffered LDS, one barrier per K-step.
// ---------------------------------------------------------------------------
__global__ __launch_bounds__(256) void gemm_wmma_kernel(
    const half_t* __restrict__ A, const half_t* __restrict__ BT,
    const float* __restrict__ bias, const float* __restrict__ resid,
    float* __restrict__ Cf, half_t* __restrict__ Ch,
    int M, int N, int K, int relu) {
  __shared__ __align__(16) _Float16 As[2][128 * 32];
  __shared__ __align__(16) _Float16 Bs[2][128 * 32];

  const int tid  = threadIdx.x;
  const int wave = tid >> 5;
  const int lane = tid & 31;
  const int g    = lane >> 4;     // half-wave group
  const int ln   = lane & 15;
  const int wm   = wave >> 1;     // 0..3 : 32-row group
  const int wn   = wave & 1;      // 0..1 : 64-col group
  const int tm   = blockIdx.y * 128;
  const int tn   = blockIdx.x * 128;

  v8f acc[2][4];
#pragma unroll
  for (int mt = 0; mt < 2; mt++)
#pragma unroll
    for (int nt = 0; nt < 4; nt++) acc[mt][nt] = (v8f){};

  // Per-thread staging pattern: 2 x v8h per tile per matrix
  const int vrow0 = tid >> 2;            // 0..63
  const int vrow1 = (tid + 256) >> 2;    // 64..127
  const int vkoff = (tid & 3) * 8;

  // Preload tile 0 into buffer 0
  *(v8h*)(&As[0][vrow0 * 32 + vkoff]) =
      *(const v8h*)(A + (size_t)(tm + vrow0) * K + vkoff);
  *(v8h*)(&As[0][vrow1 * 32 + vkoff]) =
      *(const v8h*)(A + (size_t)(tm + vrow1) * K + vkoff);
  *(v8h*)(&Bs[0][vrow0 * 32 + vkoff]) =
      *(const v8h*)(BT + (size_t)(tn + vrow0) * K + vkoff);
  *(v8h*)(&Bs[0][vrow1 * 32 + vkoff]) =
      *(const v8h*)(BT + (size_t)(tn + vrow1) * K + vkoff);
  __syncthreads();

  int buf = 0;
  for (int k0 = 0; k0 < K; k0 += 32) {
    const int nk = k0 + 32;
    v8h ra0, ra1, rb0, rb1;
    if (nk < K) {
      // Issue next tile's global loads before the WMMA chain (overlap)
      ra0 = *(const v8h*)(A + (size_t)(tm + vrow0) * K + nk + vkoff);
      ra1 = *(const v8h*)(A + (size_t)(tm + vrow1) * K + nk + vkoff);
      rb0 = *(const v8h*)(BT + (size_t)(tn + vrow0) * K + nk + vkoff);
      rb1 = *(const v8h*)(BT + (size_t)(tn + vrow1) * K + nk + vkoff);
      // Pull the tile after next into GL2 early
      if (nk + 32 < K) {
        __builtin_prefetch(A + (size_t)(tm + vrow0) * K + nk + 32 + vkoff, 0, 1);
        __builtin_prefetch(BT + (size_t)(tn + vrow0) * K + nk + 32 + vkoff, 0, 1);
      }
    }

    // A fragments (16x32) for the wave's two 16-row subtiles
    Frag16 a[2];
#pragma unroll
    for (int mt = 0; mt < 2; mt++) {
      const _Float16* ap = &As[buf][(wm * 32 + mt * 16 + ln) * 32];
      a[mt].h[0] = *(const v8h*)(ap + g * 8);
      a[mt].h[1] = *(const v8h*)(ap + 16 + g * 8);
    }
#pragma unroll
    for (int nt = 0; nt < 4; nt++) {
      Frag16 bf;  // B fragment (32x16): lane = N col, K halves at g*8 and 16+g*8
      const _Float16* bp = &Bs[buf][(wn * 64 + nt * 16 + ln) * 32];
      bf.h[0] = *(const v8h*)(bp + g * 8);
      bf.h[1] = *(const v8h*)(bp + 16 + g * 8);
#pragma unroll
      for (int mt = 0; mt < 2; mt++)
        acc[mt][nt] = wmma_f16(a[mt].v, bf.v, acc[mt][nt]);
    }

    if (nk < K) {
      const int nb = buf ^ 1;
      *(v8h*)(&As[nb][vrow0 * 32 + vkoff]) = ra0;
      *(v8h*)(&As[nb][vrow1 * 32 + vkoff]) = ra1;
      *(v8h*)(&Bs[nb][vrow0 * 32 + vkoff]) = rb0;
      *(v8h*)(&Bs[nb][vrow1 * 32 + vkoff]) = rb1;
    }
    __syncthreads();
    buf ^= 1;
  }

  // Epilogue. D layout: vgpr r -> row r + 8*g; lane%16 -> col within 16-tile.
#pragma unroll
  for (int nt = 0; nt < 4; nt++) {
    const int col = tn + wn * 64 + nt * 16 + ln;
    const float bv = bias ? bias[col] : 0.f;
#pragma unroll
    for (int mt = 0; mt < 2; mt++) {
#pragma unroll
      for (int r = 0; r < 8; r++) {
        const int row = tm + wm * 32 + mt * 16 + r + 8 * g;
        float v = acc[mt][nt][r] + bv;
        if (relu) v = fmaxf(v, 0.f);
        if (resid) v += resid[(size_t)row * N + col];
        if (Ch) Ch[(size_t)row * N + col] = (half_t)v;
        else    Cf[(size_t)row * N + col] = v;
      }
    }
  }
}

// ---------------------------------------------------------------------------
// Flash attention: one wave per (b, h, 16-query tile). Online softmax in fp32,
// S = Q K^T and O += P V via WMMA. scale = C^-0.5 = 1/32.
// ---------------------------------------------------------------------------
__global__ __launch_bounds__(32) void attn_kernel(
    const half_t* __restrict__ Q, const half_t* __restrict__ Kh,
    const half_t* __restrict__ VT, half_t* __restrict__ Oa) {
  __shared__ __align__(16) _Float16 P[16 * 32];  // P tile, [row][key] 16x32

  const int lane = threadIdx.x;
  const int g = lane >> 4, ln = lane & 15;
  const int qi = blockIdx.x;           // query tile (0..127)
  const int h  = blockIdx.y;
  const int b  = blockIdx.z;
  const int qbase = qi * 16;

  // Q fragments for the two 32-wide d-chunks of Dh=64 (kept in regs)
  Frag16 qa[2];
  {
    const half_t* qrow = Q + ((size_t)(b * SEQ + qbase + ln) * N_EMBD + h * HEADSZ);
#pragma unroll
    for (int ks = 0; ks < 2; ks++) {
      qa[ks].h[0] = *(const v8h*)(qrow + ks * 32 + g * 8);
      qa[ks].h[1] = *(const v8h*)(qrow + ks * 32 + 16 + g * 8);
    }
  }

  v8f o[4];
#pragma unroll
  for (int dt = 0; dt < 4; dt++) o[dt] = (v8f){};
  float mrow[8], lrow[8];
#pragma unroll
  for (int r = 0; r < 8; r++) { mrow[r] = -1e30f; lrow[r] = 0.f; }

  const float scale = 0.03125f;  // 1024^-0.5

  for (int kt = 0; kt <= qi; kt += 2) {   // process key tiles in pairs (32 keys)
    v8f s[2];
#pragma unroll
    for (int j = 0; j < 2; j++) {
      const int tile = kt + j;
      if (tile <= qi) {
        v8f a = (v8f){};
        const half_t* krow =
            Kh + ((size_t)(b * SEQ + tile * 16 + ln) * N_EMBD + h * HEADSZ);
#pragma unroll
        for (int ks = 0; ks < 2; ks++) {
          Frag16 kb;  // B fragment = K^T: lane = key col, halves = d at g*8 and 16+g*8
          kb.h[0] = *(const v8h*)(krow + ks * 32 + g * 8);
          kb.h[1] = *(const v8h*)(krow + ks * 32 + 16 + g * 8);
          a = wmma_f16(qa[ks].v, kb.v, a);
        }
        a *= scale;
        const int key = tile * 16 + ln;
#pragma unroll
        for (int r = 0; r < 8; r++) {
          const int qr = qbase + r + 8 * g;
          if (key > qr) a[r] = -1e30f;   // causal mask
        }
        s[j] = a;
      } else {
        v8f a = (v8f){};
        a -= 1e30f;                      // fully masked tile
        s[j] = a;
      }
    }

    // Online softmax (rows live per half-wave: width-16 shuffle reductions)
#pragma unroll
    for (int r = 0; r < 8; r++) {
      float v = fmaxf(s[0][r], s[1][r]);
      for (int msk = 1; msk < 16; msk <<= 1)
        v = fmaxf(v, __shfl_xor(v, msk, 16));
      const float mnew  = fmaxf(mrow[r], v);
      const float alpha = __expf(mrow[r] - mnew);
      const float p0 = __expf(s[0][r] - mnew);
      const float p1 = __expf(s[1][r] - mnew);
      float ps = p0 + p1;
      for (int msk = 1; msk < 16; msk <<= 1)
        ps += __shfl_xor(ps, msk, 16);
      lrow[r] = lrow[r] * alpha + ps;
      mrow[r] = mnew;
#pragma unroll
      for (int dt = 0; dt < 4; dt++) o[dt][r] *= alpha;
      // stash P (D-layout -> [row][key]) for transpose through LDS
      P[(r + 8 * g) * 32 + ln]      = (half_t)p0;
      P[(r + 8 * g) * 32 + 16 + ln] = (half_t)p1;
    }
    __syncthreads();

    // P as WMMA A-operand (16 queries x 32 keys)
    Frag16 pa;
    pa.h[0] = *(const v8h*)(&P[ln * 32 + g * 8]);
    pa.h[1] = *(const v8h*)(&P[ln * 32 + 16 + g * 8]);

    // O += P @ V over the 32 keys, four 16-wide d-tiles
#pragma unroll
    for (int dt = 0; dt < 4; dt++) {
      const half_t* vrow =
          VT + (((size_t)(b * N_HEAD + h) * HEADSZ + dt * 16 + ln) * SEQ +
                kt * 16 + g * 8);
      Frag16 vb;
      vb.h[0] = *(const v8h*)(vrow);
      vb.h[1] = *(const v8h*)(vrow + 16);
      o[dt] = wmma_f16(pa.v, vb.v, o[dt]);
    }
    __syncthreads();
  }

  // Normalize and write attention output (f16 activation buffer)
#pragma unroll
  for (int dt = 0; dt < 4; dt++) {
#pragma unroll
    for (int r = 0; r < 8; r++) {
      const float val = o[dt][r] / lrow[r];
      Oa[(size_t)(b * SEQ + qbase + r + 8 * g) * N_EMBD + h * HEADSZ + dt * 16 +
         ln] = (half_t)val;
    }
  }
}

// ---------------------------------------------------------------------------
// Host-side orchestration
// ---------------------------------------------------------------------------
extern "C" void kernel_launch(void* const* d_in, const int* in_sizes, int n_in,
                              void* d_out, int out_size, void* d_ws,
                              size_t ws_size, hipStream_t stream) {
  (void)in_sizes; (void)n_in; (void)out_size; (void)ws_size;

  const float* x     = (const float*)d_in[0];
  const float* Wq    = (const float*)d_in[1];
  const float* Wk    = (const float*)d_in[2];
  const float* Wv    = (const float*)d_in[3];
  const float* Wo    = (const float*)d_in[4];
  const float* bo    = (const float*)d_in[5];
  const float* W1    = (const float*)d_in[6];
  const float* b1    = (const float*)d_in[7];
  const float* W2    = (const float*)d_in[8];
  const float* b2    = (const float*)d_in[9];
  const float* ln1_g = (const float*)d_in[10];
  const float* ln1_b = (const float*)d_in[11];
  const float* ln2_g = (const float*)d_in[12];
  const float* ln2_b = (const float*)d_in[13];
  float* out = (float*)d_out;

  char* ws = (char*)d_ws;
  const size_t MB = 1024ull * 1024ull;
  half_t* h16    = (half_t*)(ws + 0 * MB);    // 8 MB  LN1(x) f16
  half_t* WqT    = (half_t*)(ws + 8 * MB);    // 2 MB
  half_t* WkT    = (half_t*)(ws + 10 * MB);   // 2 MB
  half_t* WvT    = (half_t*)(ws + 12 * MB);   // 2 MB
  half_t* WoT    = (half_t*)(ws + 14 * MB);   // 2 MB
  half_t* W1T    = (half_t*)(ws + 16 * MB);   // 8 MB
  half_t* W2T    = (half_t*)(ws + 24 * MB);   // 8 MB
  half_t* Q16    = (half_t*)(ws + 32 * MB);   // 8 MB
  half_t* K16    = (half_t*)(ws + 40 * MB);   // 8 MB
  half_t* V16    = (half_t*)(ws + 48 * MB);   // 8 MB
  half_t* VT16   = (half_t*)(ws + 56 * MB);   // 8 MB  [B,H,Dh,T]
  half_t* attn16 = (half_t*)(ws + 64 * MB);   // 8 MB
  float*  x1     = (float*)(ws + 72 * MB);    // 16 MB x + attn@Wo + bo
  half_t* h2_16  = (half_t*)(ws + 88 * MB);   // 8 MB  LN2(x1) f16
  half_t* ff1    = (half_t*)(ws + 96 * MB);   // 32 MB relu(h2@W1+b1) f16

  // 1) Convert + transpose weights to f16 [N][K]
  cvt_transpose_kernel<<<(N_EMBD * N_EMBD) / 256, 256, 0, stream>>>(Wq, WqT, N_EMBD, N_EMBD);
  cvt_transpose_kernel<<<(N_EMBD * N_EMBD) / 256, 256, 0, stream>>>(Wk, WkT, N_EMBD, N_EMBD);
  cvt_transpose_kernel<<<(N_EMBD * N_EMBD) / 256, 256, 0, stream>>>(Wv, WvT, N_EMBD, N_EMBD);
  cvt_transpose_kernel<<<(N_EMBD * N_EMBD) / 256, 256, 0, stream>>>(Wo, WoT, N_EMBD, N_EMBD);
  cvt_transpose_kernel<<<(N_EMBD * CFF) / 256, 256, 0, stream>>>(W1, W1T, N_EMBD, CFF);
  cvt_transpose_kernel<<<(CFF * N_EMBD) / 256, 256, 0, stream>>>(W2, W2T, CFF, N_EMBD);

  // 2) LN1 -> h16
  ln_f16_kernel<<<ROWS, 256, 0, stream>>>(x, ln1_g, ln1_b, h16);

  // 3) QKV projections (f16 out)
  dim3 gemm_blk(256);
  dim3 grid_c(N_EMBD / 128, ROWS / 128);  // (8, 32)
  gemm_wmma_kernel<<<grid_c, gemm_blk, 0, stream>>>(h16, WqT, nullptr, nullptr,
      nullptr, Q16, ROWS, N_EMBD, N_EMBD, 0);
  gemm_wmma_kernel<<<grid_c, gemm_blk, 0, stream>>>(h16, WkT, nullptr, nullptr,
      nullptr, K16, ROWS, N_EMBD, N_EMBD, 0);
  gemm_wmma_kernel<<<grid_c, gemm_blk, 0, stream>>>(h16, WvT, nullptr, nullptr,
      nullptr, V16, ROWS, N_EMBD, N_EMBD, 0);

  // 4) V -> V^T [B,H,Dh,T]
  vtranspose_kernel<<<(ROWS * N_EMBD) / 256, 256, 0, stream>>>(V16, VT16);

  // 5) Causal flash attention -> attn16
  attn_kernel<<<dim3(SEQ / 16, N_HEAD, BATCH), 32, 0, stream>>>(Q16, K16, VT16,
                                                                attn16);

  // 6) x1 = x + attn @ Wo + bo  (f32)
  gemm_wmma_kernel<<<grid_c, gemm_blk, 0, stream>>>(attn16, WoT, bo, x, x1,
      nullptr, ROWS, N_EMBD, N_EMBD, 0);

  // 7) LN2 -> h2_16
  ln_f16_kernel<<<ROWS, 256, 0, stream>>>(x1, ln2_g, ln2_b, h2_16);

  // 8) ff1 = relu(h2 @ W1 + b1)  (f16)
  gemm_wmma_kernel<<<dim3(CFF / 128, ROWS / 128), gemm_blk, 0, stream>>>(h2_16,
      W1T, b1, nullptr, nullptr, ff1, ROWS, CFF, N_EMBD, 1);

  // 9) out = x1 + ff1 @ W2 + b2  (f32)
  gemm_wmma_kernel<<<grid_c, gemm_blk, 0, stream>>>(ff1, W2T, b2, x1, out,
      nullptr, ROWS, N_EMBD, CFF, 0);
}
